// MambaBlock_50087908605955
// MI455X (gfx1250) — compile-verified, matrix operand order checked by reading, not run on hardware
//
#include <hip/hip_runtime.h>
#include <hip/hip_bf16.h>

// ---- Mamba-1 block dims (match reference) ----
#define D_MODEL 1024
#define D_INNER 2048
#define D_STATE 16
#define DT_RANK 64
#define KER 4
#define BATCH 2
#define SEQ 1024
#define BL (BATCH * SEQ)          // 2048 rows
#define XDBL_LD 128               // padded (dt_rank + 2N = 96 -> 128)

typedef __attribute__((ext_vector_type(16))) __bf16     v16bf;
typedef __attribute__((ext_vector_type(8)))  float      v8f;
typedef __attribute__((ext_vector_type(4)))  unsigned   u32x4;
typedef __attribute__((ext_vector_type(4)))  int        v4i;

// address-space-qualified pointee types (attributes must live in typedefs,
// inline attributes on cast type-names get dropped by clang)
typedef __attribute__((address_space(1))) v4i gas_v4i;   // global int4
typedef __attribute__((address_space(3))) v4i las_v4i;   // LDS int4

struct Q2 { u32x4 a, b; };

// Load one 16x32 bf16 WMMA fragment for this lane (global or LDS pointer).
// Per ISA 16-bit A layout: lanes 0-15 hold K=[0..7] (V0..V3) and K=[16..23]
// (V4..V7); lanes 16-31 hold K=[8..15] and K=[24..31]. `p` already includes
// the +8*(lane>>4) K-half offset.
__device__ __forceinline__ v16bf load_frag(const __bf16* p) {
    Q2 q;
    q.a = *(const u32x4*)(p);        // 16 bytes: 8 bf16
    q.b = *(const u32x4*)(p + 16);   // +16 elements
    return __builtin_bit_cast(v16bf, q);
}

// ---- CDNA5 async Global -> LDS copy (ASYNCcnt-tracked), b128 per lane ----
__device__ __forceinline__ void async_copy_b128(const __bf16* g, __bf16* l) {
#if __has_builtin(__builtin_amdgcn_global_load_async_to_lds_b128)
    __builtin_amdgcn_global_load_async_to_lds_b128(
        (gas_v4i*)(uintptr_t)g,
        (las_v4i*)(unsigned)(uintptr_t)l,
        0, 0);
#else
    asm volatile("global_load_async_to_lds_b128 %0, %1, off"
                 :: "v"((unsigned)(uintptr_t)l),
                    "v"((unsigned long long)(uintptr_t)g)
                 : "memory");
#endif
}

__device__ __forceinline__ void wait_async0() {
#if __has_builtin(__builtin_amdgcn_s_wait_asynccnt)
    __builtin_amdgcn_s_wait_asynccnt(0);
#else
    asm volatile("s_wait_asynccnt 0x0" ::: "memory");
#endif
}

// ---------------------------------------------------------------------------
// Generic bf16 WMMA GEMM:  C[M,N] (f32) = A[M,K] (bf16, row-major, lda)
//                                       x Bt[N,K] (bf16, row-major = B^T, ldb)
// Block = 4 waves along N; each wave owns a 64x64 tile. The block-shared
// 64x32 A-tile is double-buffered in LDS via async global->LDS loads
// (cuts A L2 traffic 4x). B streams directly via global b128 loads.
// Requires M%64==0, K%32==0, N%64==0 overall; ragged grid handled by
// clamping idle waves to a valid tile (keeps EXEC all-ones + barrier-safe).
// ---------------------------------------------------------------------------
__global__ __launch_bounds__(128) void gemm_bf16_wmma(
    const __bf16* __restrict__ A, const __bf16* __restrict__ Bt,
    float* __restrict__ C, int M, int N, int K, int lda, int ldb, int ldc) {
    const int tid   = threadIdx.x;
    const int wave  = tid >> 5;
    const int lane  = tid & 31;
    const int r     = lane & 15;
    const int kh    = lane >> 4;                 // K-half select
    const int mBase = blockIdx.x * 64;
    const int nTile = blockIdx.y * 4 + wave;
    const bool active = (nTile * 64) < N;
    const int nBase = active ? nTile * 64 : N - 64;   // clamp: stay in-bounds

    __shared__ __bf16 lA[2][64 * 32];            // 2 x 4KB double buffer

    const v8f zero = {};
    v8f acc[4][4];
#pragma unroll
    for (int i = 0; i < 4; ++i)
#pragma unroll
        for (int j = 0; j < 4; ++j) acc[i][j] = zero;

    const __bf16* bBase = Bt + (size_t)(nBase + r) * ldb + kh * 8;

    // Stage A tile [64 x 32] for K-step kk into LDS buffer `buf`:
    // 256 b128 chunks, 2 per thread.
    auto stageA = [&](int buf, int kk) {
#pragma unroll
        for (int t = 0; t < 2; ++t) {
            int c   = tid + t * 128;             // 0..255
            int row = c >> 2;
            int seg = (c & 3) * 8;               // element offset within row
            async_copy_b128(A + (size_t)(mBase + row) * lda + kk + seg,
                            &lA[buf][row * 32 + seg]);
        }
    };

    stageA(0, 0);
    wait_async0();
    __syncthreads();

    int buf = 0;
    for (int kk = 0; kk < K; kk += 32) {
        if (kk + 32 < K) stageA(buf ^ 1, kk + 32);   // prefetch next K-step

        v16bf af[4], bfr[4];
#pragma unroll
        for (int i = 0; i < 4; ++i)
            af[i] = load_frag(&lA[buf][(i * 16 + r) * 32 + kh * 8]);
#pragma unroll
        for (int j = 0; j < 4; ++j) {
            bfr[j] = load_frag(bBase + (size_t)j * 16 * ldb + kk);
            __builtin_prefetch(bBase + (size_t)j * 16 * ldb + kk + 32, 0, 0);
        }
#pragma unroll
        for (int i = 0; i < 4; ++i)
#pragma unroll
            for (int j = 0; j < 4; ++j)
                acc[i][j] = __builtin_amdgcn_wmma_f32_16x16x32_bf16(
                    false, af[i], false, bfr[j], (short)0, acc[i][j], false, false);

        wait_async0();       // my async loads for buf^1 landed
        __syncthreads();     // all waves landed theirs & finished reading buf
        buf ^= 1;
    }

    if (active) {
        // C/D layout: lane (0-15|16-31) -> N = lane&15 ; VGPR v -> M = v + 8*(lane>>4)
        const int col0 = nBase + (lane & 15);
#pragma unroll
        for (int i = 0; i < 4; ++i)
#pragma unroll
            for (int j = 0; j < 4; ++j)
#pragma unroll
                for (int v = 0; v < 8; ++v) {
                    int row = mBase + i * 16 + kh * 8 + v;
                    C[(size_t)row * ldc + col0 + j * 16] = acc[i][j][v];
                }
    }
}

// ---------------------------------------------------------------------------
// Elementwise f32 -> bf16
// ---------------------------------------------------------------------------
__global__ void f32_to_bf16(const float* __restrict__ src, __bf16* __restrict__ dst, int n) {
    int i = blockIdx.x * 256 + threadIdx.x;
    if (i < n) dst[i] = (__bf16)src[i];
}

// Transpose-convert: src f32 [Ksz x Nsz] row-major -> dst bf16 [Npad x Ksz]
// dst[n*Ksz + k] = src[k*Nsz + n], zero-padded for n in [Nsz, Npad).
__global__ void transpose_to_bf16(const float* __restrict__ src, __bf16* __restrict__ dst,
                                  int Ksz, int Nsz, int Npad) {
    int idx = blockIdx.x * 256 + threadIdx.x;
    int total = Npad * Ksz;
    if (idx >= total) return;
    int n = idx / Ksz;
    int k = idx - n * Ksz;
    float v = (n < Nsz) ? src[(size_t)k * Nsz + n] : 0.f;
    dst[idx] = (__bf16)v;
}

// ---------------------------------------------------------------------------
// Causal depthwise conv (K=4) over L + SiLU. x lives in columns [0,D_INNER) of
// xz [BL x 2*D_INNER]. Emits fp32 u (for scan) and bf16 (for x_proj GEMM).
// ---------------------------------------------------------------------------
__global__ void conv_silu(const float* __restrict__ xz, const float* __restrict__ w,
                          const float* __restrict__ bias, float* __restrict__ xconv,
                          __bf16* __restrict__ xconv_bf) {
    int idx = blockIdx.x * 256 + threadIdx.x;   // over BL * D_INNER
    if (idx >= BL * D_INNER) return;
    int d   = idx & (D_INNER - 1);
    int row = idx >> 11;                        // b*SEQ + l
    int l   = row & (SEQ - 1);
    int b   = row >> 10;
    float acc = bias[d];
#pragma unroll
    for (int k = 0; k < KER; ++k) {
        int li = l - (KER - 1) + k;
        if (li >= 0)
            acc += w[d * KER + k] * xz[((size_t)(b * SEQ + li)) * (2 * D_INNER) + d];
    }
    float s = acc / (1.f + __expf(-acc));       // SiLU
    xconv[idx]    = s;
    xconv_bf[idx] = (__bf16)s;
}

// Extract dt columns [0,64) of padded x_dbl [BL x 128] -> bf16 [BL x 64]
__global__ void extract_dt_bf16(const float* __restrict__ xdbl, __bf16* __restrict__ dt) {
    int idx = blockIdx.x * 256 + threadIdx.x;   // BL * DT_RANK
    if (idx >= BL * DT_RANK) return;
    int m = idx >> 6, k = idx & 63;
    dt[idx] = (__bf16)xdbl[(size_t)m * XDBL_LD + k];
}

// ---------------------------------------------------------------------------
// Selective scan + gating. Thread t -> (b, d, n): n = t&15, ch = t>>4,
// d = ch & 2047, b = ch >> 11. Lanes 0-15 / 16-31 of a wave32 hold the 16
// states of two channels; reduction over N via 4 xor-shuffles (stay in group).
// Fuses: softplus(dt_proj+bias), h = exp(dA)h + dBu, y = <h,C> + u*D,
// gate y * silu(res), emit bf16 for out_proj GEMM.
// ---------------------------------------------------------------------------
__global__ __launch_bounds__(256) void scan_gate(
    const float* __restrict__ delta_pre, const float* __restrict__ xconv,
    const float* __restrict__ xdbl, const float* __restrict__ xz,
    const float* __restrict__ A_log, const float* __restrict__ D_skip,
    const float* __restrict__ dt_b, __bf16* __restrict__ ygbf) {
    int t  = blockIdx.x * 256 + threadIdx.x;    // 65536 total
    int n  = t & 15;
    int ch = t >> 4;
    int d  = ch & (D_INNER - 1);
    int b  = ch >> 11;
    const float Adn  = -__expf(A_log[d * D_STATE + n]);
    const float Dd   = D_skip[d];
    const float bias = dt_b[d];
    float h = 0.f;
    for (int l = 0; l < SEQ; ++l) {
        int row    = b * SEQ + l;
        float dp   = delta_pre[(size_t)row * D_INNER + d] + bias;
        float delta = (dp > 20.f) ? dp : __logf(1.f + __expf(dp));   // softplus
        float u    = xconv[(size_t)row * D_INNER + d];
        float Bv   = xdbl[(size_t)row * XDBL_LD + DT_RANK + n];
        float Cv   = xdbl[(size_t)row * XDBL_LD + DT_RANK + D_STATE + n];
        h = __expf(delta * Adn) * h + delta * Bv * u;
        float c = h * Cv;
        c += __shfl_xor(c, 1, 32);
        c += __shfl_xor(c, 2, 32);
        c += __shfl_xor(c, 4, 32);
        c += __shfl_xor(c, 8, 32);
        if (n == 0) {
            float y  = c + u * Dd;
            float rv = xz[(size_t)row * (2 * D_INNER) + D_INNER + d];
            float g  = y * (rv / (1.f + __expf(-rv)));               // * silu(res)
            ygbf[(size_t)row * D_INNER + d] = (__bf16)g;
        }
    }
}

// ---------------------------------------------------------------------------
extern "C" void kernel_launch(void* const* d_in, const int* in_sizes, int n_in,
                              void* d_out, int out_size, void* d_ws, size_t ws_size,
                              hipStream_t stream) {
    const float* hidden    = (const float*)d_in[0];  // [2,1024,1024]
    const float* in_proj_w = (const float*)d_in[1];  // [1024,4096]
    const float* conv_w    = (const float*)d_in[2];  // [2048,1,4]
    const float* conv_b    = (const float*)d_in[3];  // [2048]
    const float* x_proj_w  = (const float*)d_in[4];  // [2048,96]
    const float* dt_proj_w = (const float*)d_in[5];  // [64,2048]
    const float* dt_proj_b = (const float*)d_in[6];  // [2048]
    const float* A_log     = (const float*)d_in[7];  // [2048,16]
    const float* D_skip    = (const float*)d_in[8];  // [2048]
    const float* out_proj_w= (const float*)d_in[9];  // [2048,1024]
    float* out = (float*)d_out;                      // [2,1024,1024]

    char* base = (char*)d_ws;
    size_t off = 0;
    auto alloc = [&](size_t bytes) -> void* {
        off = (off + 255) & ~(size_t)255;
        void* p = base + off;
        off += bytes;
        return p;
    };

    __bf16* Xbf      = (__bf16*)alloc((size_t)BL * D_MODEL * 2);        // 4 MB
    __bf16* WinT     = (__bf16*)alloc((size_t)4096 * 1024 * 2);         // 8 MB
    __bf16* XprojT   = (__bf16*)alloc((size_t)XDBL_LD * D_INNER * 2);   // 0.5 MB
    __bf16* DtprojT  = (__bf16*)alloc((size_t)D_INNER * DT_RANK * 2);   // 0.25 MB
    __bf16* WoutT    = (__bf16*)alloc((size_t)D_MODEL * D_INNER * 2);   // 4 MB
    float*  xz       = (float*)alloc((size_t)BL * 2 * D_INNER * 4);     // 32 MB
    float*  xconv    = (float*)alloc((size_t)BL * D_INNER * 4);         // 16 MB
    __bf16* xconv_bf = (__bf16*)alloc((size_t)BL * D_INNER * 2);        // 8 MB
    float*  x_dbl    = (float*)alloc((size_t)BL * XDBL_LD * 4);         // 1 MB
    __bf16* dt_bf    = (__bf16*)alloc((size_t)BL * DT_RANK * 2);        // 0.25 MB
    float*  delta_pre= (float*)alloc((size_t)BL * D_INNER * 4);         // 16 MB
    __bf16* ygbf     = (__bf16*)alloc((size_t)BL * D_INNER * 2);        // 8 MB

    // 1) operand conversions (bf16, weights transposed for B-fragment loads)
    {
        int n = BL * D_MODEL;
        f32_to_bf16<<<(n + 255) / 256, 256, 0, stream>>>(hidden, Xbf, n);
    }
    { int tot = 4096 * 1024;
      transpose_to_bf16<<<(tot + 255) / 256, 256, 0, stream>>>(in_proj_w, WinT, 1024, 4096, 4096); }
    { int tot = XDBL_LD * D_INNER;
      transpose_to_bf16<<<(tot + 255) / 256, 256, 0, stream>>>(x_proj_w, XprojT, D_INNER, 96, XDBL_LD); }
    { int tot = D_INNER * DT_RANK;
      transpose_to_bf16<<<(tot + 255) / 256, 256, 0, stream>>>(dt_proj_w, DtprojT, DT_RANK, D_INNER, D_INNER); }
    { int tot = D_MODEL * D_INNER;
      transpose_to_bf16<<<(tot + 255) / 256, 256, 0, stream>>>(out_proj_w, WoutT, D_INNER, D_MODEL, D_MODEL); }

    // 2) in_proj: xz[2048,4096] = X[2048,1024] @ Win
    gemm_bf16_wmma<<<dim3(BL / 64, 4096 / 256), 128, 0, stream>>>(
        Xbf, WinT, xz, BL, 4096, D_MODEL, D_MODEL, D_MODEL, 4096);

    // 3) causal depthwise conv + SiLU
    { int n = BL * D_INNER;
      conv_silu<<<(n + 255) / 256, 256, 0, stream>>>(xz, conv_w, conv_b, xconv, xconv_bf); }

    // 4) x_proj: x_dbl[2048,128pad] = xconv @ Wx  (N=96 zero-padded to 128)
    gemm_bf16_wmma<<<dim3(BL / 64, 1), 128, 0, stream>>>(
        xconv_bf, XprojT, x_dbl, BL, XDBL_LD, D_INNER, D_INNER, D_INNER, XDBL_LD);

    // 5) dt slice -> bf16; dt_proj: delta_pre[2048,2048] = dt @ Wdt
    { int n = BL * DT_RANK;
      extract_dt_bf16<<<(n + 255) / 256, 256, 0, stream>>>(x_dbl, dt_bf); }
    gemm_bf16_wmma<<<dim3(BL / 64, D_INNER / 256), 128, 0, stream>>>(
        dt_bf, DtprojT, delta_pre, BL, D_INNER, DT_RANK, DT_RANK, DT_RANK, D_INNER);

    // 6) selective scan + SiLU gate (N=16 states across lane groups)
    scan_gate<<<(BATCH * D_INNER * D_STATE) / 256, 256, 0, stream>>>(
        delta_pre, xconv, x_dbl, xz, A_log, D_skip, dt_proj_b, ygbf);

    // 7) out_proj: out[2048,1024] = ygated @ Wout
    gemm_bf16_wmma<<<dim3(BL / 64, D_MODEL / 256), 128, 0, stream>>>(
        ygbf, WoutT, out, BL, D_MODEL, D_INNER, D_INNER, D_INNER, D_MODEL);
}
